// WanModelMoe_5239860101493
// MI455X (gfx1250) — matrix operand order repaired and checked by reading, not compiled, not run on hardware
//
#include <hip/hip_runtime.h>

typedef __attribute__((ext_vector_type(16))) _Float16 v16h;
typedef __attribute__((ext_vector_type(4)))  _Float16 v4h;
typedef __attribute__((ext_vector_type(8)))  float    v8f;
typedef __attribute__((ext_vector_type(4)))  float    v4f;

#define TOK  32768   // B*S tokens
#define DIM  1024    // D
#define OUT  1024    // O
#define NE   4       // experts
#define KBLK 32      // DIM/32 k-steps
#define BM   32      // tokens per block (2 WMMA m-tiles)
#define BN   128     // output cols per block (8 waves x 16)

// ---------------------------------------------------------------------------
// Pre-pass: convert W [E, D, O] fp32 -> f16 in WMMA B-operand layout.
// B (32x16 KxN) per lane, contiguous k order used here (not AMD's table order): lanes 0-15: N=lane, halves = K 0..15;
//                         lanes 16-31: N=lane-16, halves = K 16..31.
// Packed so each lane loads 32 contiguous bytes per (e, kb, n):
//   Bh[((e*KBLK + kb)*OUT + n)*32 + (k % 32)]
// ---------------------------------------------------------------------------
__global__ void __launch_bounds__(256) moe_convert_w(const float* __restrict__ W,
                                                     _Float16* __restrict__ Bh) {
    int idx = blockIdx.x * 256 + threadIdx.x;   // NE*DIM*OUT, coalesced read
    int n   = idx & (OUT - 1);
    int ek  = idx >> 10;
    int k   = ek & (DIM - 1);
    int e   = ek >> 10;
    int kb  = k >> 5;
    size_t dst = ((size_t)((e * KBLK + kb) * OUT + n)) * 32 + (k & 31);
    Bh[dst] = (_Float16)W[idx];
}

__device__ inline float sel_expert(int e, float v0, float v1, float v2, float v3) {
    float r = v3;
    r = (e == 2) ? v2 : r;
    r = (e == 1) ? v1 : r;
    r = (e == 0) ? v0 : r;
    return r;
}

#define WMMA_F16(A, B, C) \
    __builtin_amdgcn_wmma_f32_16x16x32_f16(false, (A), false, (B), (short)0, (C), false, false)

// ---------------------------------------------------------------------------
// Main kernel. Block: 256 threads (8 waves), tile = 32 tokens x 128 cols.
// Each wave: 2 m-tiles x 16 cols x 4 experts = 8 f32 accumulators.
// A staged once in LDS (f16, A-operand lane order); B register-double-buffered
// behind one affine pointer (constant stride, immediate expert offsets).
// ---------------------------------------------------------------------------
__global__ void __launch_bounds__(256) moe_wmma_kernel(
        const float*    __restrict__ x,     // [TOK, DIM] fp32
        const float*    __restrict__ ew,    // [TOK, 2]   fp32 top-k weights
        const int*      __restrict__ ti,    // [TOK, 2]   int  top-k indices
        const _Float16* __restrict__ Bh,    // packed f16 W
        const float*    __restrict__ bias,  // [NE, OUT]  fp32
        float*          __restrict__ out)   // [TOK, OUT] fp32
{
    // A in LDS, per (mt, kb) tile in exact per-lane operand order:
    // halves index = (((mt*KBLK + kb)*32 + lane) << 4) + h   (32B per lane)
    __shared__ _Float16 Alds[BM * DIM];     // 64 KB

    const int tid     = threadIdx.x;
    const int lane    = tid & 31;
    const int wave    = tid >> 5;
    const int tokbase = blockIdx.x * BM;
    const int nbase   = blockIdx.y * BN + wave * 16;
    const int hi      = lane >> 4;
    const int ncol    = lane & 15;

    // ---- Prologue: cooperative A stage, fp32 -> f16 into LDS ----
    // Thread owns one 4-wide k-column (k = tid*4) across all 32 token rows;
    // all layout math is loop-invariant, so the loop is load+cvt+ds_store only.
    {
        const int k   = tid << 2;                    // 0..1020
        const int kp  = k & 31;
        const int hiK = kp >> 4;
        const int h   = kp & 15;                     // half slot (mult of 4)
        const float* xrow  = x + (size_t)tokbase * DIM + k;
        _Float16*    abase = &Alds[((((k >> 5) * 32) + hiK * 16) << 4) + h];
#pragma unroll
        for (int i = 0; i < BM; ++i) {               // token row = i
            v4f f = *(const v4f*)(xrow + i * DIM);
            v4h hv;
            hv[0] = (_Float16)f[0]; hv[1] = (_Float16)f[1];
            hv[2] = (_Float16)f[2]; hv[3] = (_Float16)f[3];
            const int mt = i >> 4, ml = i & 15;
            *(v4h*)&abase[(mt * (KBLK * 32) + ml) << 4] = hv;
        }
    }
    __syncthreads();

    // ---- B: one affine pointer; expert offsets are immediates (e * 2MB) ----
    const _Float16* bbase = Bh + ((size_t)(nbase + ncol)) * 32 + (size_t)hi * 16;
#define BSTEP ((size_t)OUT * 32)            // halves per k-step slot
#define BEXP  ((size_t)KBLK * OUT * 32)     // halves per expert (2MB bytes)

    v8f c[2][NE];
#pragma unroll
    for (int mt = 0; mt < 2; ++mt)
#pragma unroll
        for (int e = 0; e < NE; ++e) c[mt][e] = (v8f){};

    // Preload k-step 0 into the current buffer.
    v16h bc0 = *(const v16h*)(bbase + 0 * BEXP);
    v16h bc1 = *(const v16h*)(bbase + 1 * BEXP);
    v16h bc2 = *(const v16h*)(bbase + 2 * BEXP);
    v16h bc3 = *(const v16h*)(bbase + 3 * BEXP);

    const _Float16* bp = bbase + BSTEP;     // points at k-step kb+1

#pragma unroll 4
    for (int kb = 0; kb < KBLK - 1; ++kb) {
        // Prefetch k-step kb+2 toward near cache (speculative; OOB tail is
        // silently dropped / lands in unused ws).
        __builtin_prefetch(bp + 0 * BEXP + BSTEP, 0, 3);
        __builtin_prefetch(bp + 1 * BEXP + BSTEP, 0, 3);
        __builtin_prefetch(bp + 2 * BEXP + BSTEP, 0, 3);
        __builtin_prefetch(bp + 3 * BEXP + BSTEP, 0, 3);

        // Issue next k-step's B loads; they retire under the WMMAs below.
        v16h bn0 = *(const v16h*)(bp + 0 * BEXP);
        v16h bn1 = *(const v16h*)(bp + 1 * BEXP);
        v16h bn2 = *(const v16h*)(bp + 2 * BEXP);
        v16h bn3 = *(const v16h*)(bp + 3 * BEXP);

        // A tiles for both m-tiles from LDS (immediate ds offsets).
        v16h a0 = *(const v16h*)&Alds[((0 * KBLK + kb) * 32 + lane) << 4];
        v16h a1 = *(const v16h*)&Alds[((1 * KBLK + kb) * 32 + lane) << 4];

        c[0][0] = WMMA_F16(a0, bc0, c[0][0]);
        c[1][0] = WMMA_F16(a1, bc0, c[1][0]);
        c[0][1] = WMMA_F16(a0, bc1, c[0][1]);
        c[1][1] = WMMA_F16(a1, bc1, c[1][1]);
        c[0][2] = WMMA_F16(a0, bc2, c[0][2]);
        c[1][2] = WMMA_F16(a1, bc2, c[1][2]);
        c[0][3] = WMMA_F16(a0, bc3, c[0][3]);
        c[1][3] = WMMA_F16(a1, bc3, c[1][3]);

        bc0 = bn0; bc1 = bn1; bc2 = bn2; bc3 = bn3;
        bp += BSTEP;
    }

    // Peeled final k-step (kb = KBLK-1): no next-load, no wrap arithmetic.
    {
        v16h a0 = *(const v16h*)&Alds[((0 * KBLK + (KBLK - 1)) * 32 + lane) << 4];
        v16h a1 = *(const v16h*)&Alds[((1 * KBLK + (KBLK - 1)) * 32 + lane) << 4];
        c[0][0] = WMMA_F16(a0, bc0, c[0][0]);
        c[1][0] = WMMA_F16(a1, bc0, c[1][0]);
        c[0][1] = WMMA_F16(a0, bc1, c[0][1]);
        c[1][1] = WMMA_F16(a1, bc1, c[1][1]);
        c[0][2] = WMMA_F16(a0, bc2, c[0][2]);
        c[1][2] = WMMA_F16(a1, bc2, c[1][2]);
        c[0][3] = WMMA_F16(a0, bc3, c[0][3]);
        c[1][3] = WMMA_F16(a1, bc3, c[1][3]);
    }
#undef BSTEP
#undef BEXP

    // ---- Combine: out[t,col] = sum_k w[t,k] * (acc[idx[t,k]] + bias[idx[t,k],col])
    // C layout: lanes 0-15 -> N=lane, VGPR r -> M=r; lanes 16-31 -> M=8+r.
    const int col = nbase + ncol;
    const float bb0 = bias[0 * OUT + col];
    const float bb1 = bias[1 * OUT + col];
    const float bb2 = bias[2 * OUT + col];
    const float bb3 = bias[3 * OUT + col];

#pragma unroll
    for (int mt = 0; mt < 2; ++mt) {
#pragma unroll
        for (int r = 0; r < 8; ++r) {
            const int t  = tokbase + mt * 16 + r + hi * 8;
            const float w0 = ew[t * 2 + 0];
            const float w1 = ew[t * 2 + 1];
            const int   i0 = ti[t * 2 + 0];
            const int   i1 = ti[t * 2 + 1];
            const float e0 = c[mt][0][r] + bb0;
            const float e1 = c[mt][1][r] + bb1;
            const float e2 = c[mt][2][r] + bb2;
            const float e3 = c[mt][3][r] + bb3;
            const float g0 = sel_expert(i0, e0, e1, e2, e3);
            const float g1 = sel_expert(i1, e0, e1, e2, e3);
            out[(size_t)t * OUT + col] = w0 * g0 + w1 * g1;
        }
    }
}

extern "C" void kernel_launch(void* const* d_in, const int* in_sizes, int n_in,
                              void* d_out, int out_size, void* d_ws, size_t ws_size,
                              hipStream_t stream) {
    const float* x    = (const float*)d_in[0];   // [4,8192,1024]
    const float* ew   = (const float*)d_in[1];   // [4,8192,2]
    const int*   ti   = (const int*)  d_in[2];   // [4,8192,2]
    const float* W    = (const float*)d_in[3];   // [4,1024,1024]
    const float* bias = (const float*)d_in[4];   // [4,1024]
    float*       out  = (float*)d_out;           // [4,8192,1024]
    _Float16*    Bh   = (_Float16*)d_ws;         // 8 MB packed f16 weights

    // 1) pack/convert expert weights into WMMA B-operand layout (f16)
    moe_convert_w<<<dim3((NE * DIM * OUT) / 256), dim3(256), 0, stream>>>(W, Bh);

    // 2) fused dense-expert GEMM + top-k gather/combine
    moe_wmma_kernel<<<dim3(TOK / BM, OUT / BN), dim3(256), 0, stream>>>(
        x, ew, ti, Bh, bias, out);
}
